// AttentionLayer_13202729468349
// MI455X (gfx1250) — matrix operand order repaired and checked by reading, not compiled, not run on hardware
//
#include <hip/hip_runtime.h>
#include <math.h>

// Problem constants (fixed by the reference).
#define BB   8
#define CC   256
#define HH   128
#define WW   128
#define SS   (HH * WW)      // 16384
#define KK   19

#define ROWS  128           // spatial rows (s) per block
#define PITCH 264           // bf16 elements per LDS row (256 + 8 pad, 16B-aligned rows)

typedef __attribute__((ext_vector_type(16))) __bf16 v16bf;
typedef __attribute__((ext_vector_type(8)))  __bf16 v8bf;
typedef __attribute__((ext_vector_type(2)))  __bf16 v2bf;
typedef __attribute__((ext_vector_type(8)))  float  v8f;

// Native CDNA5 v_tanh_f32 if the toolchain exposes it; branchless fallback.
__device__ __forceinline__ float fast_tanh(float x) {
#if __has_builtin(__builtin_amdgcn_tanhf)
  return __builtin_amdgcn_tanhf(x);
#else
  float e = __expf(x + x);                 // v_exp_f32
  return __fdividef(e - 1.0f, e + 1.0f);   // fast rcp path
#endif
}

// ---------------------------------------------------------------------------
// Prologue 1: img_fc_w (f32, [256,256]) -> bf16.
// ---------------------------------------------------------------------------
__global__ __launch_bounds__(256) void conv_weights_bf16(
    const float* __restrict__ w, __bf16* __restrict__ wbf) {
  int i = (blockIdx.x * 256 + threadIdx.x) * 4;
  float4 v = *(const float4*)(w + i);
  wbf[i + 0] = (__bf16)v.x;
  wbf[i + 1] = (__bf16)v.y;
  wbf[i + 2] = (__bf16)v.z;
  wbf[i + 3] = (__bf16)v.w;
}

// ---------------------------------------------------------------------------
// Prologue 2: bias2  = img_fc_b + kp_fc_w @ kp_proj_b + kp_fc_b   [256]
//             M2     = kp_fc_w @ kp_proj_w                        [256,19]
//             kp_pix = pixel index per (b,k), -1 if invisible     [8*19]
// ---------------------------------------------------------------------------
__global__ __launch_bounds__(256) void prep_small(
    const float* __restrict__ kp_feat,    // [B,K,3]
    const float* __restrict__ kp_proj_w,  // [K,K]
    const float* __restrict__ kp_proj_b,  // [K]
    const float* __restrict__ kp_fc_w,    // [C,K]
    const float* __restrict__ kp_fc_b,    // [C]
    const float* __restrict__ img_fc_b,   // [C]
    float* __restrict__ bias2,            // [C]
    float* __restrict__ M2,               // [C,K]
    int* __restrict__ kp_pix) {           // [B*K]
  const int o = threadIdx.x;  // one block, 256 threads
  float vb = kp_fc_b[o] + img_fc_b[o];
  for (int j = 0; j < KK; ++j) vb += kp_fc_w[o * KK + j] * kp_proj_b[j];
  bias2[o] = vb;
  for (int k = 0; k < KK; ++k) {
    float m = 0.0f;
    for (int j = 0; j < KK; ++j) m += kp_fc_w[o * KK + j] * kp_proj_w[j * KK + k];
    M2[o * KK + k] = m;
  }
  if (o < BB * KK) {
    float kx  = kp_feat[o * 3 + 0];
    float ky  = kp_feat[o * 3 + 1];
    float vis = kp_feat[o * 3 + 2];
    int x = (int)fminf(fmaxf(kx * (1.0f / (float)WW), 0.0f), (float)(WW - 1));
    int y = (int)fminf(fmaxf(ky * (1.0f / (float)HH), 0.0f), (float)(HH - 1));
    kp_pix[o] = (vis > 0.0f) ? (y * WW + x) : -1;
  }
}

// ---------------------------------------------------------------------------
// Main fused kernel (dense path, NO keypoint logic):
//   Y = tanh( X @ W^T + bias2 )
//   score = sigmoid( Y @ attn_w + attn_b );  out = X_f32 * score
// GEMM runs on v_wmma_f32_16x16x32_bf16; X staged transposed in LDS as bf16.
// The ~152 keypoint-touched pixels are overwritten exactly by fixup_kp after.
// ---------------------------------------------------------------------------
__global__ __launch_bounds__(256) void attn_main(
    const float* __restrict__ img,       // [B,C,S]
    const float* __restrict__ attn_w,    // [C]
    const float* __restrict__ attn_b,    // [1]
    const __bf16* __restrict__ Wbf,      // [C,C] bf16 (row o, contiguous c)
    const float* __restrict__ bias2,     // [C]
    float* __restrict__ out) {           // [B,C,S]
  __shared__ __attribute__((aligned(16))) __bf16 Xt[ROWS * PITCH];  // [r][c] bf16
  __shared__ float scores[ROWS];

  const int tid = threadIdx.x;
  const int b   = blockIdx.y;
  const int s0  = blockIdx.x * ROWS;
  const size_t base = (size_t)b * CC * SS;

  // ---- Phase 1: coalesced load of [C x ROWS] tile, bf16-convert, transpose.
  // Each task handles a channel PAIR so LDS writes are packed 4-byte v2bf.
  for (int i = 0; i < 16; ++i) {
    int flat = i * 256 + tid;          // 4096 pair-tasks
    int c2 = flat >> 5;                // channel pair 0..127
    int r4 = (flat & 31) << 2;         // 4 rows per task
    const size_t g = base + (size_t)(2 * c2) * SS + s0 + r4;
    const float4 va = *(const float4*)(img + g);
    const float4 vb = *(const float4*)(img + g + SS);
    v2bf p0 = {(__bf16)va.x, (__bf16)vb.x};
    v2bf p1 = {(__bf16)va.y, (__bf16)vb.y};
    v2bf p2 = {(__bf16)va.z, (__bf16)vb.z};
    v2bf p3 = {(__bf16)va.w, (__bf16)vb.w};
    *(v2bf*)&Xt[(r4 + 0) * PITCH + 2 * c2] = p0;
    *(v2bf*)&Xt[(r4 + 1) * PITCH + 2 * c2] = p1;
    *(v2bf*)&Xt[(r4 + 2) * PITCH + 2 * c2] = p2;
    *(v2bf*)&Xt[(r4 + 3) * PITCH + 2 * c2] = p3;
  }
  __syncthreads();

  const int lane = tid & 31;
  const int warp = tid >> 5;       // 8 waves; wave w owns rows [w*16, w*16+16)
  const int hf   = lane >> 4;      // lane half (per ISA 16-bit A/B layouts)
  const int l16  = lane & 15;
  const int rowA = warp * 16 + l16;

  union ABu { v16bf v; v8bf h[2]; };

  v8f attnacc = {0.f, 0.f, 0.f, 0.f, 0.f, 0.f, 0.f, 0.f};
  const float attn_bias = attn_b[0];

  for (int ot = 0; ot < 16; ++ot) {          // 16 output-channel tiles of 16
    const int   o    = ot * 16 + l16;        // this lane's output channel (B-col N)
    const float bias = bias2[o];
    const float aw   = attn_w[o];

    v8f acc = {0.f, 0.f, 0.f, 0.f, 0.f, 0.f, 0.f, 0.f};
    for (int kt = 0; kt < 8; ++kt) {         // K = 256 in steps of 32
      // A fragment: lane half 0 -> K {0..15}, half 1 -> K {16..31}
      ABu a;
      const int ca = kt * 32 + (hf ? 16 : 0);
      a.h[0] = *(const v8bf*)&Xt[rowA * PITCH + ca];
      a.h[1] = *(const v8bf*)&Xt[rowA * PITCH + ca + 8];
      // B fragment: column N = l16, K = kt*32 + (hf?16:0) .. +15, contiguous bf16
      const int kb = kt * 32 + (hf ? 16 : 0);
      const v16bf bfr = *(const v16bf*)(Wbf + (size_t)o * CC + kb);
      acc = __builtin_amdgcn_wmma_f32_16x16x32_bf16(
          false, a.v, false, bfr, (short)0, acc, false, false);
    }

    // Epilogue on the 16x16 f32 accumulator: VGPR j, half hf -> row m = j + 8*hf
    for (int j = 0; j < 8; ++j) {
      attnacc[j] += aw * fast_tanh(acc[j] + bias);
    }
  }

  // Reduce the attention dot across the 16 lanes of each half, then sigmoid.
  for (int j = 0; j < 8; ++j) {
    float v = attnacc[j];
    v += __shfl_xor(v, 1);
    v += __shfl_xor(v, 2);
    v += __shfl_xor(v, 4);
    v += __shfl_xor(v, 8);
    if (l16 == 0) {
      float sc = 1.0f / (1.0f + __expf(-(v + attn_bias)));
      scores[warp * 16 + j + (hf << 3)] = sc;
    }
  }
  __syncthreads();

  // ---- Phase 4: attended = img * score, fully coalesced float4 traffic
  for (int i = 0; i < 32; ++i) {
    int flat = i * 256 + tid;
    int c  = flat >> 5;
    int r4 = (flat & 31) << 2;
    size_t gidx = base + (size_t)c * SS + s0 + r4;
    const float4 v = *(const float4*)(img + gidx);
    float4 ov;
    ov.x = v.x * scores[r4 + 0];
    ov.y = v.y * scores[r4 + 1];
    ov.z = v.z * scores[r4 + 2];
    ov.w = v.w * scores[r4 + 3];
    *(float4*)(out + gidx) = ov;
  }
}

// ---------------------------------------------------------------------------
// Fix-up: for each visible keypoint pixel (<=152 total), recompute the full
// 256-channel result in f32 (including all M2 corrections for keypoints that
// share this pixel) and overwrite out[b,:,s]. Idempotent across duplicates.
// ---------------------------------------------------------------------------
__global__ __launch_bounds__(256) void fixup_kp(
    const float* __restrict__ img,       // [B,C,S]
    const float* __restrict__ img_fc_w,  // [C,C] f32
    const float* __restrict__ attn_w,    // [C]
    const float* __restrict__ attn_b,    // [1]
    const float* __restrict__ bias2,     // [C]
    const float* __restrict__ M2,        // [C,K]
    const int* __restrict__ kp_pix,      // [B*K]
    float* __restrict__ out) {           // [B,C,S]
  const int bk = blockIdx.x;             // 0 .. B*K-1
  const int b  = bk / KK;
  const int s  = kp_pix[bk];
  if (s < 0) return;

  __shared__ float xs[CC];
  __shared__ float red[CC];
  const int o = threadIdx.x;
  const size_t base = (size_t)b * CC * SS;

  xs[o] = img[base + (size_t)o * SS + s];

  // Sum corrections from every keypoint of this batch that maps to pixel s.
  float corr = 0.0f;
  for (int k = 0; k < KK; ++k)
    if (kp_pix[b * KK + k] == s) corr += M2[o * KK + k];
  __syncthreads();

  float val = bias2[o] + corr;
  for (int c = 0; c < CC; ++c) val += img_fc_w[o * CC + c] * xs[c];
  red[o] = attn_w[o] * fast_tanh(val);
  __syncthreads();

  for (int off = 128; off > 0; off >>= 1) {
    if (o < off) red[o] += red[o + off];
    __syncthreads();
  }
  float score = 1.0f / (1.0f + __expf(-(red[0] + attn_b[0])));
  out[base + (size_t)o * SS + s] = xs[o] * score;
}

// ---------------------------------------------------------------------------
// Host launcher
// ---------------------------------------------------------------------------
extern "C" void kernel_launch(void* const* d_in, const int* in_sizes, int n_in,
                              void* d_out, int out_size, void* d_ws, size_t ws_size,
                              hipStream_t stream) {
  const float* img       = (const float*)d_in[0];  // [B,C,H,W]
  const float* kp_feat   = (const float*)d_in[1];  // [B,K,3]
  const float* img_fc_w  = (const float*)d_in[2];  // [C,C]
  const float* img_fc_b  = (const float*)d_in[3];  // [C]
  const float* kp_proj_w = (const float*)d_in[4];  // [K,K]
  const float* kp_proj_b = (const float*)d_in[5];  // [K]
  const float* kp_fc_w   = (const float*)d_in[6];  // [C,K]
  const float* kp_fc_b   = (const float*)d_in[7];  // [C]
  const float* attn_w    = (const float*)d_in[8];  // [1,C]
  const float* attn_b    = (const float*)d_in[9];  // [1]

  char* ws = (char*)d_ws;
  __bf16* Wbf    = (__bf16*)(ws);            // 131072 B
  float*  bias2  = (float*)(ws + 131072);    //   1024 B
  float*  M2     = (float*)(ws + 132096);    //  19456 B
  int*    kp_pix = (int*)  (ws + 151552);    //    608 B

  conv_weights_bf16<<<64, 256, 0, stream>>>(img_fc_w, Wbf);
  prep_small<<<1, 256, 0, stream>>>(kp_feat, kp_proj_w, kp_proj_b,
                                    kp_fc_w, kp_fc_b, img_fc_b,
                                    bias2, M2, kp_pix);

  dim3 grid(SS / ROWS, BB);  // (128, 8)
  attn_main<<<grid, 256, 0, stream>>>(img, attn_w, attn_b,
                                      Wbf, bias2, (float*)d_out);

  fixup_kp<<<BB * KK, 256, 0, stream>>>(img, img_fc_w, attn_w, attn_b,
                                        bias2, M2, kp_pix, (float*)d_out);
}
